// ScaledDotProductAttention_32856499814621
// MI455X (gfx1250) — compile-verified, matrix operand order checked by reading, not run on hardware
//
#include <hip/hip_runtime.h>
#include <stdint.h>

typedef __attribute__((ext_vector_type(16))) _Float16 v16h;
typedef __attribute__((ext_vector_type(8)))  float    v8f;

#define WMMA_F16(A, B, C) \
  __builtin_amdgcn_wmma_f32_16x16x32_f16(false, (A), false, (B), (short)0, (C), false, false)

// Problem shape (fixed by the reference)
constexpr int Bb = 4, Hh = 16, Ss = 2048, Dd = 64;
constexpr int QB    = 16;                  // q-rows per wave
constexpr int NQB   = Bb * Hh * (Ss / QB); // 8192 q-blocks
constexpr int WAVES = 4;                   // waves per workgroup

// Load 16 f16 A/B-operand elements for one lane:
// elements 0..7  <- p[0..7]   (K = koff + 8*hsel + 0..7)
// elements 8..15 <- p[16..23] (K = koff + 16 + 8*hsel + 0..7)
__device__ inline v16h load_row16(const float* __restrict__ p) {
  v16h r;
#pragma unroll
  for (int i = 0; i < 8; ++i) {
    r[i]     = (_Float16)p[i];
    r[8 + i] = (_Float16)p[i + 16];
  }
  return r;
}

__device__ inline float redmax16(float v) {
  v = fmaxf(v, __shfl_xor(v, 1, 32));
  v = fmaxf(v, __shfl_xor(v, 2, 32));
  v = fmaxf(v, __shfl_xor(v, 4, 32));
  v = fmaxf(v, __shfl_xor(v, 8, 32));
  return v;
}
__device__ inline float redsum16(float v) {
  v += __shfl_xor(v, 1, 32);
  v += __shfl_xor(v, 2, 32);
  v += __shfl_xor(v, 4, 32);
  v += __shfl_xor(v, 8, 32);
  return v;
}

__global__ __launch_bounds__(WAVES * 32)
void sdpa_wmma_kernel(const float* __restrict__ Q,
                      const float* __restrict__ K,
                      const float* __restrict__ V,
                      const uint8_t* __restrict__ M,   // bool mask, 1 byte/elem
                      float* __restrict__ ctx,         // (B,H,S,Dd)
                      float* __restrict__ attn)        // (B,H,S,S)
{
  __shared__ float   pbuf[WAVES][2][16][17];          // P-tile transpose staging
  __shared__ __align__(16) uint8_t mbuf[WAVES][16][Ss]; // mask strip (32KB/wave)

  const int tid  = threadIdx.x;
  const int wave = tid >> 5;
  const int lane = tid & 31;

  const int qb = blockIdx.x * WAVES + wave;
  if (qb >= NQB) return;

  const int bh   = qb / (Ss / QB);          // fused (b*H + h)
  const int q0   = (qb % (Ss / QB)) * QB;   // first q row of this block
  const int ln16 = lane & 15;               // A: row m ; B/C/D: col n
  const int hsel = lane >> 4;               // half-wave select
  const int mlo  = 8 * hsel;                // C/D rows for this lane: m = r + mlo

  const size_t baseF = (size_t)bh * Ss * Dd;  // Q/K/V feature-plane base
  const size_t baseM = (size_t)bh * Ss * Ss;  // mask/attn plane base

  // ---- Kick off async copy of this wave's contiguous 32KB mask strip ----
  // rows q0..q0+15 of the (S,S) mask plane are contiguous in memory.
  {
    const uint8_t* msrc = M + baseM + (size_t)q0 * Ss;
    uint8_t* mdst = &mbuf[wave][0][0];
#pragma unroll 1
    for (int i = lane; i < (16 * Ss) / 16; i += 32) {   // 2048 x 16B chunks
      uint32_t laddr = (uint32_t)(uintptr_t)(mdst + (size_t)i * 16);
      uint64_t gaddr = (uint64_t)(uintptr_t)(msrc + (size_t)i * 16);
      asm volatile("global_load_async_to_lds_b128 %0, %1, off"
                   :: "v"(laddr), "v"(gaddr) : "memory");
    }
  }

  // Q tile A-operands (held for the whole kernel)
  const float* qrow = Q + baseF + (size_t)(q0 + ln16) * Dd + mlo;
  const v16h aq0 = load_row16(qrow);        // K-features 0..31
  const v16h aq1 = load_row16(qrow + 32);   // K-features 32..63

  // raw QK^T tile (16x16 accumulators, C/D layout, unscaled)
  auto qk_tile = [&](int k0) -> v8f {
    const float* krow = K + baseF + (size_t)(k0 + ln16) * Dd + mlo;
    v16h b0 = load_row16(krow);
    v16h b1 = load_row16(krow + 32);
    v8f c = {};
    c = WMMA_F16(aq0, b0, c);
    c = WMMA_F16(aq1, b1, c);
    return c;
  };

  // scaled + masked scores (mask bytes from LDS strip)
  auto masked_scores = [&](int k0) -> v8f {
    v8f c = qk_tile(k0);
    v8f s;
#pragma unroll
    for (int r = 0; r < 8; ++r) {
      float sv = c[r] * 0.125f;  // 1/sqrt(64)
      uint8_t mk = mbuf[wave][r + mlo][k0 + ln16];
      s[r] = mk ? -1.0e9f : sv;
    }
    return s;
  };

  // ---- Pass A1: per-lane running max (unmasked upper bound; no shuffles) ----
  float mxl[8];
#pragma unroll
  for (int r = 0; r < 8; ++r) mxl[r] = -3.0e38f;
#pragma unroll 1
  for (int kb = 0; kb < Ss / 16; ++kb) {
    v8f c = qk_tile(kb * 16);
#pragma unroll
    for (int r = 0; r < 8; ++r) mxl[r] = fmaxf(mxl[r], c[r] * 0.125f);
  }
  float mx[8];
#pragma unroll
  for (int r = 0; r < 8; ++r) mx[r] = redmax16(mxl[r]);

  // mask strip must be in LDS before pass A2
  asm volatile("s_wait_asynccnt 0" ::: "memory");

  // ---- Pass A2: per-lane running sum of exp(s - mx) (no shuffles in loop) ----
  float sml[8];
#pragma unroll
  for (int r = 0; r < 8; ++r) sml[r] = 0.0f;
#pragma unroll 1
  for (int kb = 0; kb < Ss / 16; ++kb) {
    v8f s = masked_scores(kb * 16);
#pragma unroll
    for (int r = 0; r < 8; ++r) sml[r] += __expf(s[r] - mx[r]);
  }
  float inv[8];
#pragma unroll
  for (int r = 0; r < 8; ++r) inv[r] = 1.0f / fmaxf(redsum16(sml[r]), 1.0e-30f);

  // ---- Pass B: write normalized attn, accumulate context = P.V -------------
  v8f acc[4];
#pragma unroll
  for (int n0 = 0; n0 < 4; ++n0) acc[n0] = (v8f){};

#pragma unroll 1
  for (int kp = 0; kp < Ss / 32; ++kp) {
#pragma unroll
    for (int half = 0; half < 2; ++half) {
      const int k0 = kp * 32 + half * 16;
      v8f s = masked_scores(k0);
#pragma unroll
      for (int r = 0; r < 8; ++r) {
        float p = __expf(s[r] - mx[r]) * inv[r];
        attn[baseM + (size_t)(q0 + r + mlo) * Ss + (k0 + ln16)] = p;
        pbuf[wave][half][r + mlo][ln16] = p;  // stage for lane transpose
      }
    }

    // Rebuild P (16x32) in A-operand layout from LDS (in-wave DS is in-order)
    v16h ap;
#pragma unroll
    for (int i = 0; i < 8; ++i) {
      ap[i]     = (_Float16)pbuf[wave][0][ln16][mlo + i];  // K = 8*hsel + i
      ap[8 + i] = (_Float16)pbuf[wave][1][ln16][mlo + i];  // K = 16 + 8*hsel + i
    }

    const int k0 = kp * 32;
#pragma unroll
    for (int n0 = 0; n0 < 4; ++n0) {
      v16h bv;  // V block (32 x 16), B-operand layout
#pragma unroll
      for (int i = 0; i < 8; ++i) {
        const int kl0 = mlo + i;        // element i
        const int kl1 = 16 + mlo + i;   // element 8+i
        bv[i]     = (_Float16)V[baseF + (size_t)(k0 + kl0) * Dd + n0 * 16 + ln16];
        bv[8 + i] = (_Float16)V[baseF + (size_t)(k0 + kl1) * Dd + n0 * 16 + ln16];
      }
      acc[n0] = WMMA_F16(ap, bv, acc[n0]);
    }
  }

  // ---- Write context -------------------------------------------------------
#pragma unroll
  for (int n0 = 0; n0 < 4; ++n0)
#pragma unroll
    for (int r = 0; r < 8; ++r)
      ctx[baseF + (size_t)(q0 + r + mlo) * Dd + n0 * 16 + ln16] = acc[n0][r];
}

extern "C" void kernel_launch(void* const* d_in, const int* in_sizes, int n_in,
                              void* d_out, int out_size, void* d_ws, size_t ws_size,
                              hipStream_t stream) {
  (void)in_sizes; (void)n_in; (void)out_size; (void)d_ws; (void)ws_size;
  const float*   Q = (const float*)d_in[0];
  const float*   K = (const float*)d_in[1];
  const float*   V = (const float*)d_in[2];
  const uint8_t* M = (const uint8_t*)d_in[3];  // jnp.bool_ mask, 1 byte/elem

  float* out  = (float*)d_out;
  float* ctx  = out;                                   // (B,H,S,Dd) first
  float* attn = out + (size_t)Bb * Hh * Ss * Dd;       // then (B,H,S,S)

  dim3 grid(NQB / WAVES);   // 2048
  dim3 block(WAVES * 32);   // 128 threads = 4 waves
  sdpa_wmma_kernel<<<grid, block, 0, stream>>>(Q, K, V, M, ctx, attn);
}
